// AttentionHead_8083128451565
// MI455X (gfx1250) — matrix operand order repaired and checked by reading, not compiled, not run on hardware
//
#include <hip/hip_runtime.h>
#include <cstdint>

#define HIDDEN   1024
#define HEAD_DIM 64
#define BATCH    8
#define SEQ      2048

typedef __attribute__((ext_vector_type(16))) __bf16 v16bf;
typedef __attribute__((ext_vector_type(8)))  float  v8f;
typedef __attribute__((ext_vector_type(4)))  float  v4f;
typedef __attribute__((ext_vector_type(4)))  unsigned int v4u;
typedef __attribute__((ext_vector_type(2)))  unsigned int v2u;

union FragBF {
    v16bf  v;
    __bf16 e[16];
    v4u    u[2];
};

static __device__ __forceinline__ v8f wmma_bf16(v16bf a, v16bf b, v8f c) {
    // D = A(16x32 bf16) * B(32x16 bf16) + C(16x16 f32)
    return __builtin_amdgcn_wmma_f32_16x16x32_bf16(false, a, false, b,
                                                   (short)0, c, false, false);
}

// CDNA5 async global->LDS copy (16B per lane), tracked by ASYNCcnt.
static __device__ __forceinline__ void async_copy16(uint32_t lds_off, const void* g) {
    asm volatile("global_load_async_to_lds_b128 %0, %1, off"
                 :: "v"(lds_off), "v"((unsigned long long)(uintptr_t)g)
                 : "memory");
}
static __device__ __forceinline__ void wait_async0() {
    asm volatile("s_wait_asynccnt 0" ::: "memory");
}
static __device__ __forceinline__ void lds_wave_fence() {
    asm volatile("s_wait_dscnt 0" ::: "memory");
    __builtin_amdgcn_wave_barrier();
}

// ---------------------------------------------------------------------------
// Kernel 0: one-shot fp32 -> bf16 conversion of the three 64x1024 W matrices.
// grid = (64, 3), block = 256; each thread converts 4 elements.
// ---------------------------------------------------------------------------
__global__ void __launch_bounds__(256)
wconv_kernel(const float* __restrict__ Wq, const float* __restrict__ Wk,
             const float* __restrict__ Wv,
             unsigned short* __restrict__ owq, unsigned short* __restrict__ owk,
             unsigned short* __restrict__ owv)
{
    const float* W; __bf16* O;
    if (blockIdx.y == 0)      { W = Wq; O = (__bf16*)owq; }
    else if (blockIdx.y == 1) { W = Wk; O = (__bf16*)owk; }
    else                      { W = Wv; O = (__bf16*)owv; }

    const int i = (blockIdx.x * 256 + threadIdx.x) * 4;   // < 64*1024
    v4f f = *(const v4f*)(W + i);
    union { __bf16 e[4]; v2u u; } t;
#pragma unroll
    for (int j = 0; j < 4; ++j) t.e[j] = (__bf16)f[j];
    *(v2u*)(O + i) = t.u;
}

// ---------------------------------------------------------------------------
// Kernel 1: q/k/v projections.  X(16x1024 f32) * W^T(1024x64 bf16) + b -> bf16.
// One wave per 16-row tile; K reduced in 32-wide bf16 WMMA steps.
// blockDim = 128 (4 waves); grid = (B*S/16/4, 3)
// ---------------------------------------------------------------------------
__global__ void __launch_bounds__(128)
proj_kernel(const float* __restrict__ xq, const float* __restrict__ xk,
            const float* __restrict__ xv,
            const unsigned short* __restrict__ wqb,
            const unsigned short* __restrict__ wkb,
            const unsigned short* __restrict__ wvb,
            const float* __restrict__ bq, const float* __restrict__ bk,
            const float* __restrict__ bv,
            unsigned short* __restrict__ oq, unsigned short* __restrict__ ok,
            unsigned short* __restrict__ ov)
{
    const int lane = threadIdx.x & 31;
    const int wave = threadIdx.x >> 5;
    const int col  = lane & 15;   // N / row-within-A index
    const int hb   = lane >> 4;   // lane half (0/1)

    const float* X; const __bf16* W; const float* bb; __bf16* O;
    if (blockIdx.y == 0)      { X = xq; W = (const __bf16*)wqb; bb = bq; O = (__bf16*)oq; }
    else if (blockIdx.y == 1) { X = xk; W = (const __bf16*)wkb; bb = bk; O = (__bf16*)ok; }
    else                      { X = xv; W = (const __bf16*)wvb; bb = bv; O = (__bf16*)ov; }

    const int rowtile = blockIdx.x * 4 + wave;          // 0..1023
    const long arow   = (long)rowtile * 16 + col;       // A-matrix row = lane%16
    const float* xrow = X + arow * HIDDEN;

    v8f c[4] = {};                                      // 16x64 f32 accumulators

    for (int kt = 0; kt < HIDDEN; kt += 32) {
        // A fragment: element j -> K = kt + (j/8)*16 + hb*8 + (j%8)
        FragBF a;
        {
            __align__(16) float tf[16];
            const float* ap0 = xrow + kt + hb * 8;       // j = 0..7
            const float* ap1 = xrow + kt + 16 + hb * 8;  // j = 8..15
            *(v4f*)(tf + 0)  = *(const v4f*)(ap0);
            *(v4f*)(tf + 4)  = *(const v4f*)(ap0 + 4);
            *(v4f*)(tf + 8)  = *(const v4f*)(ap1);
            *(v4f*)(tf + 12) = *(const v4f*)(ap1 + 4);
#pragma unroll
            for (int j = 0; j < 16; ++j) a.e[j] = (__bf16)tf[j];
        }
#pragma unroll
        for (int n = 0; n < 4; ++n) {
            // B fragment: N = n*16+col (output dim), K = kt + (e/8)*16 + hb*8 + (e%8)
            FragBF b;
            const __bf16* wp = W + (long)(n * 16 + col) * HIDDEN + kt + hb * 8;
            b.u[0] = *(const v4u*)(wp);
            b.u[1] = *(const v4u*)(wp + 16);
            c[n] = wmma_bf16(a.v, b.v, c[n]);
        }
    }

    // C layout: lane holds rows M = hb*8 + r, column N = col (within n-tile)
#pragma unroll
    for (int n = 0; n < 4; ++n) {
        const int   d    = n * 16 + col;
        const float bias = bb[d];
#pragma unroll
        for (int r = 0; r < 8; ++r) {
            const long orow = (long)rowtile * 16 + hb * 8 + r;
            O[orow * HEAD_DIM + d] = (__bf16)(c[n][r] + bias);
        }
    }
}

// ---------------------------------------------------------------------------
// Kernel 2: flash attention. One wave per 16-row query tile, 32-key steps.
// blockDim = 128 (4 waves); grid = (SEQ/64, BATCH)
// ---------------------------------------------------------------------------
__global__ void __launch_bounds__(128)
attn_kernel(const unsigned short* __restrict__ qws,
            const unsigned short* __restrict__ kws,
            const unsigned short* __restrict__ vws,
            const float* __restrict__ relb, const int* __restrict__ mask,
            float* __restrict__ out)
{
    __shared__ __align__(16) __bf16 kTile[32 * 64];       // [key][dim]
    __shared__ __align__(16) __bf16 vTileT[64 * 32];      // [dim][key]
    __shared__ __align__(16) __bf16 pStage[4][16 * 32];   // per-wave [row][key]

    const int lane = threadIdx.x & 31;
    const int wave = threadIdx.x >> 5;
    const int col  = lane & 15;
    const int hb   = lane >> 4;
    const int b    = blockIdx.y;
    const int qtile = blockIdx.x * 64 + wave * 16;
    const long sbase = (long)b * SEQ;                     // row base in ws arrays

    const __bf16* qp = (const __bf16*)qws;
    const __bf16* kp = (const __bf16*)kws;
    const __bf16* vp = (const __bf16*)vws;

    // Q as two A fragments (K-dim = head dim 0..31 / 32..63)
    FragBF a0, a1;
    {
        const __bf16* qr = qp + (sbase + qtile + col) * HEAD_DIM;
        a0.u[0] = *(const v4u*)(qr + hb * 16);
        a0.u[1] = *(const v4u*)(qr + 8 + hb * 16);
        a1.u[0] = *(const v4u*)(qr + 32 + hb * 16);
        a1.u[1] = *(const v4u*)(qr + 40 + hb * 16);
    }

    float m[8], l[8];
    v8f o[4] = {};
#pragma unroll
    for (int r = 0; r < 8; ++r) { m[r] = -3.0e38f; l[r] = 0.0f; }

    const float scale = 0.125f;                           // 1/sqrt(64)

    for (int kt = 0; kt < SEQ; kt += 32) {
        __syncthreads();                                  // protect LDS reuse
        // Async copy of K tile (32 keys x 64 dims bf16 = 4KB contiguous)
        {
            const char* ksrc = (const char*)(kp + (sbase + kt) * HEAD_DIM);
            const uint32_t lbase = (uint32_t)(uintptr_t)&kTile[0];
            const int t = threadIdx.x;
            async_copy16(lbase + t * 16,         ksrc + t * 16);
            async_copy16(lbase + (t + 128) * 16, ksrc + (t + 128) * 16);
        }
        // V tile, transposed into LDS for contiguous B-fragment reads
        {
            const int t   = threadIdx.x;
            const int key = t >> 2;
            const int d0  = (t & 3) * 16;
            const __bf16* vsrc = vp + (sbase + kt + key) * HEAD_DIM + d0;
            __align__(16) __bf16 tv[16];
            *(v4u*)(tv + 0) = *(const v4u*)(vsrc);
            *(v4u*)(tv + 8) = *(const v4u*)(vsrc + 8);
#pragma unroll
            for (int j = 0; j < 16; ++j) vTileT[(d0 + j) * 32 + key] = tv[j];
        }
        wait_async0();
        __syncthreads();

        // S = Q * K^T : two 16x16 tiles (keys kt..kt+15, kt+16..kt+31)
        v8f s0 = {}, s1 = {};
        {
            FragBF bk;
            const __bf16* kl0 = &kTile[col * 64 + hb * 16];
            bk.u[0] = *(const v4u*)(kl0);      bk.u[1] = *(const v4u*)(kl0 + 8);
            s0 = wmma_bf16(a0.v, bk.v, s0);
            bk.u[0] = *(const v4u*)(kl0 + 32); bk.u[1] = *(const v4u*)(kl0 + 40);
            s0 = wmma_bf16(a1.v, bk.v, s0);
            const __bf16* kl1 = &kTile[(16 + col) * 64 + hb * 16];
            bk.u[0] = *(const v4u*)(kl1);      bk.u[1] = *(const v4u*)(kl1 + 8);
            s1 = wmma_bf16(a0.v, bk.v, s1);
            bk.u[0] = *(const v4u*)(kl1 + 32); bk.u[1] = *(const v4u*)(kl1 + 40);
            s1 = wmma_bf16(a1.v, bk.v, s1);
        }

        // bias + scale + mask (single streaming pass over relative_biases)
        const int mk0 = mask[b * SEQ + kt + col];
        const int mk1 = mask[b * SEQ + kt + 16 + col];
        const long rb = (sbase + qtile + hb * 8) * SEQ + kt + col;
#pragma unroll
        for (int r = 0; r < 8; ++r) {
            const float bi0 = relb[rb + (long)r * SEQ];
            const float bi1 = relb[rb + (long)r * SEQ + 16];
            float x0 = (s0[r] + bi0) * scale;
            float x1 = (s1[r] + bi1) * scale;
            s0[r] = mk0 ? x0 : -3.0e38f;
            s1[r] = mk1 ? x1 : -3.0e38f;
        }

        // online softmax: row stats live replicated across the 16-lane half
#pragma unroll
        for (int r = 0; r < 8; ++r) {
            float vmax = fmaxf(s0[r], s1[r]);
            vmax = fmaxf(vmax, __shfl_xor(vmax, 1));
            vmax = fmaxf(vmax, __shfl_xor(vmax, 2));
            vmax = fmaxf(vmax, __shfl_xor(vmax, 4));
            vmax = fmaxf(vmax, __shfl_xor(vmax, 8));
            const float mn    = fmaxf(m[r], vmax);
            const float alpha = __expf(m[r] - mn);
            m[r] = mn;
            const float p0 = __expf(s0[r] - mn);
            const float p1 = __expf(s1[r] - mn);
            s0[r] = p0; s1[r] = p1;
            float rs = p0 + p1;
            rs += __shfl_xor(rs, 1);
            rs += __shfl_xor(rs, 2);
            rs += __shfl_xor(rs, 4);
            rs += __shfl_xor(rs, 8);
            l[r] = l[r] * alpha + rs;
#pragma unroll
            for (int n = 0; n < 4; ++n) o[n][r] *= alpha;
        }

        // stage P (C layout) -> LDS -> reload as A fragment (per-wave region)
        __bf16* ps = &pStage[wave][0];
#pragma unroll
        for (int r = 0; r < 8; ++r) {
            const int row = hb * 8 + r;
            ps[row * 32 + col]      = (__bf16)s0[r];
            ps[row * 32 + 16 + col] = (__bf16)s1[r];
        }
        lds_wave_fence();
        FragBF pa;
        {
            const __bf16* pr = ps + col * 32 + hb * 16;
            pa.u[0] = *(const v4u*)(pr);
            pa.u[1] = *(const v4u*)(pr + 8);
        }

        // O += P(16x32) * V(32x64)
#pragma unroll
        for (int n = 0; n < 4; ++n) {
            FragBF vb;
            const __bf16* vl = &vTileT[(n * 16 + col) * 32 + hb * 16];
            vb.u[0] = *(const v4u*)(vl);
            vb.u[1] = *(const v4u*)(vl + 8);
            o[n] = wmma_bf16(pa.v, vb.v, o[n]);
        }
    }

    // normalize and store fp32 output
#pragma unroll
    for (int r = 0; r < 8; ++r) {
        const float inv  = 1.0f / l[r];
        const long  orow = sbase + qtile + hb * 8 + r;
#pragma unroll
        for (int n = 0; n < 4; ++n)
            out[orow * HEAD_DIM + n * 16 + col] = o[n][r] * inv;
    }
}

// ---------------------------------------------------------------------------
extern "C" void kernel_launch(void* const* d_in, const int* in_sizes, int n_in,
                              void* d_out, int out_size, void* d_ws, size_t ws_size,
                              hipStream_t stream) {
    const float* q    = (const float*)d_in[0];
    const float* k    = (const float*)d_in[1];
    const float* v    = (const float*)d_in[2];
    const float* relb = (const float*)d_in[3];
    const int*   mask = (const int*)d_in[4];
    const float* Wq   = (const float*)d_in[5];
    const float* bq   = (const float*)d_in[6];
    const float* Wk   = (const float*)d_in[7];
    const float* bk   = (const float*)d_in[8];
    const float* Wv   = (const float*)d_in[9];
    const float* bv   = (const float*)d_in[10];
    float* out = (float*)d_out;

    const size_t projN = (size_t)BATCH * SEQ * HEAD_DIM;   // 1M elements
    const size_t wN    = (size_t)HEAD_DIM * HIDDEN;        // 64K elements
    unsigned short* qws = (unsigned short*)d_ws;
    unsigned short* kws = qws + projN;
    unsigned short* vws = kws + projN;
    unsigned short* wqb = vws + projN;
    unsigned short* wkb = wqb + wN;
    unsigned short* wvb = wkb + wN;

    dim3 g0(64, 3);                       // 64*256*4 = 65536 elements per W
    wconv_kernel<<<g0, 256, 0, stream>>>(Wq, Wk, Wv, wqb, wkb, wvb);

    dim3 g1((BATCH * SEQ / 16) / 4, 3);   // 256 x 3 blocks, 4 waves each
    proj_kernel<<<g1, 128, 0, stream>>>(q, k, v, wqb, wkb, wvb, bq, bk, bv,
                                        qws, kws, vws);

    dim3 g2(SEQ / 64, BATCH);             // 32 x 8 blocks, 4 waves each
    attn_kernel<<<g2, 128, 0, stream>>>(qws, kws, vws, relb, mask, out);
}
